// PennyLaneHead_69183333204503
// MI455X (gfx1250) — compile-verified, matrix operand order checked by reading, not run on hardware
//
#include <hip/hip_runtime.h>
#include <cmath>

typedef __attribute__((ext_vector_type(16))) __bf16 vbf16;
typedef __attribute__((ext_vector_type(8)))  float  v8f;

#define NQ    10
#define DEPTH 6

// =====================================================================
// Kernel A0: prepack W_proj into WMMA-ready B fragments (bf16 hi/lo),
// zero-padded N 10->16, layout [kstep][lane][16 elems] (32 B / fragment).
// One work-item per (kstep, lane) fragment.
// =====================================================================
__global__ __launch_bounds__(256) void prepack_w_kernel(
    const float* __restrict__ Wp, __bf16* __restrict__ bhp,
    __bf16* __restrict__ blp, int F)
{
  const int idx = blockIdx.x * 256 + threadIdx.x;   // (F/32)*32 fragments
  const int ksteps = F >> 5;
  if (idx >= ksteps * 32) return;
  const int lane = idx & 31;
  const int l16  = lane & 15;
  const int half = lane >> 4;
  const int kb   = (idx >> 5) * 32 + half * 8;
  const bool valid = (l16 < NQ);
  const float* wrow = Wp + (size_t)(valid ? l16 : 0) * F;
#pragma unroll
  for (int e = 0; e < 16; ++e) {
    const int k = kb + ((e < 8) ? e : (e + 8));     // e>=8 -> kb+16+(e-8)
    float g = valid ? wrow[k] : 0.0f;
    __bf16 h = (__bf16)g;
    bhp[(size_t)idx * 16 + e] = h;
    blp[(size_t)idx * 16 + e] = (__bf16)(g - (float)h);
  }
}

// =====================================================================
// Kernel A: projection GEMM  angles = tanh(x @ Wp^T) * pi/2
// 16x16x32 bf16 WMMA; fp32 via bf16 hi/lo split (3 WMMAs / K-step).
// A fragments from x via float4 loads; B fragments preloaded packed.
// =====================================================================
__global__ __launch_bounds__(256) void proj_wmma_kernel(
    const float* __restrict__ x,
    const vbf16* __restrict__ bhp, const vbf16* __restrict__ blp,
    float* __restrict__ angles, int F)
{
  const int wave = threadIdx.x >> 5;
  const int lane = threadIdx.x & 31;
  const int tile = blockIdx.x * 8 + wave;   // M tile index
  const int half = lane >> 4;               // K half select
  const int l16  = lane & 15;
  const int arow = tile * 16 + l16;         // A-matrix row for this lane
  const float4* x4 = (const float4*)(x + (size_t)arow * F);

  v8f acc = {};
  const int ksteps = F >> 5;
#pragma unroll 2
  for (int ks = 0; ks < ksteps; ++ks) {
    const int kb = ks * 32 + half * 8;      // multiple of 8
    const int q0 = kb >> 2;                 // float4 index
    float4 a0 = x4[q0];                     // K kb   .. kb+3
    float4 a1 = x4[q0 + 1];                 // K kb+4 .. kb+7
    float4 a2 = x4[q0 + 4];                 // K kb+16.. kb+19
    float4 a3 = x4[q0 + 5];                 // K kb+20.. kb+23
    const float fa[16] = {a0.x, a0.y, a0.z, a0.w, a1.x, a1.y, a1.z, a1.w,
                          a2.x, a2.y, a2.z, a2.w, a3.x, a3.y, a3.z, a3.w};
    vbf16 ah, al;
#pragma unroll
    for (int j = 0; j < 16; ++j) {
      __bf16 h = (__bf16)fa[j];
      ah[j] = h;
      al[j] = (__bf16)(fa[j] - (float)h);
    }
    vbf16 bhf = bhp[ks * 32 + lane];        // one aligned 32B fragment load
    vbf16 blf = blp[ks * 32 + lane];
    // hi*hi + lo*hi + hi*lo  (lo*lo ~ 2^-16 rel, dropped)
    acc = __builtin_amdgcn_wmma_f32_16x16x32_bf16(false, ah, false, bhf, (short)0, acc, false, false);
    acc = __builtin_amdgcn_wmma_f32_16x16x32_bf16(false, al, false, bhf, (short)0, acc, false, false);
    acc = __builtin_amdgcn_wmma_f32_16x16x32_bf16(false, ah, false, blf, (short)0, acc, false, false);
  }
  // C/D layout: VGPR r, lanes 0-15 -> M=r, lanes 16-31 -> M=r+8, N=lane&15
#pragma unroll
  for (int r = 0; r < 8; ++r) {
    int row = tile * 16 + r + half * 8;
    angles[(size_t)row * 16 + l16] = tanhf(acc[r]) * 1.5707963267948966f;
  }
}

// =====================================================================
// Kernel B: register-resident 10-qubit statevector sim, 1 batch / wave32.
// Index bits [9:5] = lane, [4:0] = register slot. Qubit q -> bit p = 9-q.
// p<5: in-lane (compile-time reg indices); p>=5: shfl_xor cross-lane.
// =====================================================================
__device__ __forceinline__ float xorshfl(float v, int m) {
  return __shfl_xor(v, m, 32);
}

struct U2 { float a_r, a_i, b_r, b_i, c_r, c_i, d_r, d_i; }; // U00 U01 U10 U11

template<int Q>
__device__ __forceinline__ void apply1q(float* re, float* im, int lane, const U2& u) {
  constexpr int P = 9 - Q;
  if constexpr (P < 5) {
#pragma unroll
    for (int t0 = 0; t0 < 32; ++t0) {
      if ((t0 & (1 << P)) == 0) {
        const int t1 = t0 | (1 << P);
        float xr = re[t0], xi = im[t0], yr = re[t1], yi = im[t1];
        re[t0] = u.a_r*xr - u.a_i*xi + u.b_r*yr - u.b_i*yi;
        im[t0] = u.a_r*xi + u.a_i*xr + u.b_r*yi + u.b_i*yr;
        re[t1] = u.c_r*xr - u.c_i*xi + u.d_r*yr - u.d_i*yi;
        im[t1] = u.c_r*xi + u.c_i*xr + u.d_r*yi + u.d_i*yr;
      }
    }
  } else {
    constexpr int m = 1 << (P - 5);
    const bool hi = (lane & m) != 0;
    const float ar = hi ? u.d_r : u.a_r, ai = hi ? u.d_i : u.a_i; // * mine
    const float br = hi ? u.c_r : u.b_r, bi = hi ? u.c_i : u.b_i; // * partner
#pragma unroll
    for (int t = 0; t < 32; ++t) {
      float pr = xorshfl(re[t], m), pi = xorshfl(im[t], m);
      float mr = re[t], mi = im[t];
      re[t] = ar*mr - ai*mi + br*pr - bi*pi;
      im[t] = ar*mi + ai*mr + br*pi + bi*pr;
    }
  }
}

template<int C, int T>
__device__ __forceinline__ void cnotg(float* re, float* im, int lane) {
  constexpr int PC = 9 - C, PT = 9 - T;
  if constexpr (PC < 5 && PT < 5) {
    // pure register rename: zero instructions after SROA
#pragma unroll
    for (int t = 0; t < 32; ++t) {
      if ((t & (1 << PC)) && !(t & (1 << PT))) {
        const int t1 = t | (1 << PT);
        float tr = re[t]; re[t] = re[t1]; re[t1] = tr;
        float ti = im[t]; im[t] = im[t1]; im[t1] = ti;
      }
    }
  } else if constexpr (PC >= 5 && PT < 5) {
    const bool c1 = ((lane >> (PC - 5)) & 1) != 0;
#pragma unroll
    for (int t = 0; t < 32; ++t) {
      if ((t & (1 << PT)) == 0) {
        const int t1 = t | (1 << PT);
        float r0 = re[t], r1 = re[t1];
        re[t] = c1 ? r1 : r0;  re[t1] = c1 ? r0 : r1;
        float i0 = im[t], i1 = im[t1];
        im[t] = c1 ? i1 : i0;  im[t1] = c1 ? i0 : i1;
      }
    }
  } else if constexpr (PC < 5 && PT >= 5) {
    constexpr int m = 1 << (PT - 5);
#pragma unroll
    for (int t = 0; t < 32; ++t) {
      if (t & (1 << PC)) {      // compile-time per slot; both lanes of pair agree
        re[t] = xorshfl(re[t], m);
        im[t] = xorshfl(im[t], m);
      }
    }
  } else {
    constexpr int m = 1 << (PT - 5);
    const bool c1 = ((lane >> (PC - 5)) & 1) != 0;
#pragma unroll
    for (int t = 0; t < 32; ++t) {
      float sr = xorshfl(re[t], m), si = xorshfl(im[t], m);
      re[t] = c1 ? sr : re[t];
      im[t] = c1 ? si : im[t];
    }
  }
}

template<int L, int Q>
__device__ __forceinline__ void rot_chain(float* re, float* im, int lane,
                                          const float* __restrict__ w) {
  if constexpr (Q < NQ) {
    const float phi = w[(L*NQ + Q)*3 + 0];
    const float th  = w[(L*NQ + Q)*3 + 1];
    const float om  = w[(L*NQ + Q)*3 + 2];
    float sh, ch; __sincosf(0.5f * th, &sh, &ch);
    float sa, ca; __sincosf(0.5f * (phi + om), &sa, &ca);
    float sb, cb; __sincosf(0.5f * (phi - om), &sb, &cb);
    U2 u;
    u.a_r =  ca*ch; u.a_i = -sa*ch;   // ep*c
    u.b_r = -cb*sh; u.b_i = -sb*sh;   // -conj(em)*s
    u.c_r =  cb*sh; u.c_i = -sb*sh;   // em*s
    u.d_r =  ca*ch; u.d_i =  sa*ch;   // conj(ep)*c
    apply1q<Q>(re, im, lane, u);
    rot_chain<L, Q + 1>(re, im, lane, w);
  }
}

template<int L, int I>
__device__ __forceinline__ void cnot_chain(float* re, float* im, int lane) {
  if constexpr (I < NQ) {
    constexpr int R = (L % (NQ - 1)) + 1;
    cnotg<I, (I + R) % NQ>(re, im, lane);
    cnot_chain<L, I + 1>(re, im, lane);
  }
}

template<int L>
__device__ __forceinline__ void layers(float* re, float* im, int lane,
                                       const float* __restrict__ w) {
  if constexpr (L < DEPTH) {
    rot_chain<L, 0>(re, im, lane, w);
    cnot_chain<L, 0>(re, im, lane);
    layers<L + 1>(re, im, lane, w);
  }
}

__global__ __launch_bounds__(256) void vqc_sim_kernel(
    const float* __restrict__ angles,   // (B,16) padded
    const float* __restrict__ weights,  // (DEPTH,NQ,3)
    const float* __restrict__ W_out,    // (10,NQ)
    const float* __restrict__ b_out,    // (10)
    float* __restrict__ out)            // (B,10)
{
  const int lane = threadIdx.x & 31;
  const int b    = blockIdx.x * 8 + (threadIdx.x >> 5);
  const float* ap = angles + (size_t)b * 16;

  // --- RY-encoded product state (all-real): amp = prod(bit ? sin : cos) ---
  float cq[NQ], sq[NQ];
#pragma unroll
  for (int q = 0; q < NQ; ++q) __sincosf(0.5f * ap[q], &sq[q], &cq[q]);

  float lf = 1.0f;                         // lane-bit factor (qubits 0..4)
#pragma unroll
  for (int q = 0; q < 5; ++q)
    lf *= ((lane >> (4 - q)) & 1) ? sq[q] : cq[q];

  float re[32], im[32];
#pragma unroll
  for (int t = 0; t < 32; ++t) {
    float a = lf;
#pragma unroll
    for (int q = 5; q < NQ; ++q)          // compile-time bit pick after unroll
      a *= ((t >> (9 - q)) & 1) ? sq[q] : cq[q];
    re[t] = a; im[t] = 0.0f;
  }

  // --- 6 strongly-entangling layers, fully unrolled ---
  layers<0>(re, im, lane, weights);

  // --- <Z_q> measurement ---
  float tot = 0.0f;
  float zin[5] = {0.f, 0.f, 0.f, 0.f, 0.f};
#pragma unroll
  for (int t = 0; t < 32; ++t) {
    float pt = re[t]*re[t] + im[t]*im[t];
    tot += pt;
#pragma unroll
    for (int p = 0; p < 5; ++p)
      zin[p] += ((t >> p) & 1) ? -pt : pt;
  }
  float z[NQ];
#pragma unroll
  for (int q = 5; q < NQ; ++q) z[q] = zin[9 - q];
#pragma unroll
  for (int q = 0; q < 5; ++q)
    z[q] = (((lane >> (4 - q)) & 1) != 0) ? -tot : tot;
#pragma unroll
  for (int q = 0; q < NQ; ++q) {
#pragma unroll
    for (int s = 1; s < 32; s <<= 1) z[q] += xorshfl(z[q], s);
  }

  // --- fused output head: lanes 0..9 each produce one class ---
  if (lane < 10) {
    float acc = b_out[lane];
#pragma unroll
    for (int q = 0; q < NQ; ++q) acc += W_out[lane*NQ + q] * z[q];
    out[(size_t)b * 10 + lane] = acc;
  }
}

// =====================================================================
extern "C" void kernel_launch(void* const* d_in, const int* in_sizes, int n_in,
                              void* d_out, int out_size, void* d_ws, size_t ws_size,
                              hipStream_t stream) {
  const float* x       = (const float*)d_in[0];   // (B,F)
  const float* W_proj  = (const float*)d_in[1];   // (NQ,F)
  const float* weights = (const float*)d_in[2];   // (DEPTH,NQ,3)
  const float* W_out   = (const float*)d_in[3];   // (10,NQ)
  const float* b_out   = (const float*)d_in[4];   // (10)
  float* out = (float*)d_out;

  const int F = in_sizes[1] / NQ;                 // 1024
  const int B = in_sizes[0] / F;                  // 8192
  const int ksteps = F >> 5;                      // 32

  // workspace layout: angles (B*16 f32) | W bf16-hi frags | W bf16-lo frags
  char* ws = (char*)d_ws;
  float*  angles = (float*)ws;
  __bf16* bhp    = (__bf16*)(ws + (size_t)B * 16 * sizeof(float));
  __bf16* blp    = bhp + (size_t)ksteps * 32 * 16;

  const int frag = ksteps * 32;                   // 1024 fragments
  prepack_w_kernel<<<(frag + 255) / 256, 256, 0, stream>>>(W_proj, bhp, blp, F);

  const int mtiles     = B / 16;                  // 512 waves
  const int projBlocks = (mtiles + 7) / 8;        // 8 waves / block
  proj_wmma_kernel<<<projBlocks, 256, 0, stream>>>(
      x, (const vbf16*)bhp, (const vbf16*)blp, angles, F);

  const int simBlocks = (B + 7) / 8;              // 1 batch / wave, 8 waves / block
  vqc_sim_kernel<<<simBlocks, 256, 0, stream>>>(angles, weights, W_out, b_out, out);
}
